// GNN_basis_2_79680233276083
// MI455X (gfx1250) — compile-verified, matrix operand order checked by reading, not run on hardware
//
#include <hip/hip_runtime.h>

// ---------------------------------------------------------------------------
// GNN forward for MI455X (gfx1250, wave32).
// Edge-MLP GEMMs dominate (~120 GFLOP). Strategy: bf16 activations so the
// per-layer edge ping-pong buffers (2 x 64 MiB) stay L2-resident (192 MB L2),
// v_wmma_f32_16x16x32_bf16 with fp32 accumulation, fp32 BN stats / residual.
// Weights are staged per-block into LDS (320 KB/WGP) so B-fragments come from
// ds_load_b128 shared by all 8 waves; only the A stream hits global memory.
// ---------------------------------------------------------------------------

typedef __bf16 bf16;
typedef __attribute__((ext_vector_type(16))) __bf16 v16bf;
typedef __attribute__((ext_vector_type(8)))  __bf16 v8bf;
typedef __attribute__((ext_vector_type(8)))  float   v8f;

static constexpr int cB = 16, cN = 128, cE = 16384, cEMB = 64;
static constexpr int cBN = cB * cN;                   // 2048 node rows
static constexpr long long cBE = (long long)cB * cE;  // 262144 edge rows

#define CDIV(a, b) (((a) + (b) - 1) / (b))

__device__ __forceinline__ float swishf(float x) { return x / (1.f + __expf(-x)); }

// Load a 16-element bf16 fragment slice per the CDNA5 16-bit A/B layout:
// 8 contiguous bf16 at p, 8 more at p+16 (K and K+16 slabs).
__device__ __forceinline__ v16bf load_frag(const bf16* __restrict__ p) {
  v8bf lo = *(const v8bf*)(p);
  v8bf hi = *(const v8bf*)(p + 16);
  return __builtin_shufflevector(lo, hi, 0, 1, 2, 3, 4, 5, 6, 7,
                                         8, 9, 10, 11, 12, 13, 14, 15);
}

// ---------------------------------------------------------------------------
// Generic GEMM: Out[M,N] = act(A[M,K] @ W[K,N] + bias), A bf16 row-major,
// Wt = W^T bf16 [N,K]. Block = 8 waves covering 128 rows x 64 cols; the
// 64 x K weight strip is staged once into LDS (row stride K+8 to spread
// banks) and shared by all waves. Each wave: 16x64 strip, one A fragment
// feeds 4 WMMAs per K-step. Requires K%32==0, N%64==0, M%128==0.
// ---------------------------------------------------------------------------
template <bool OUT_BF16, bool SWISH>
__global__ void wmma_gemm_kernel(const bf16* __restrict__ A, const bf16* __restrict__ Wt,
                                 const float* __restrict__ bias, void* __restrict__ Out,
                                 int M, int N, int K) {
  __shared__ bf16 sWt[64 * (256 + 8)];  // max K=256, +8 bf16 row pad (16 B)
  const int tid = threadIdx.x;
  const int lane = tid & 31;
  const int wave = tid >> 5;
  const int nGroups = N >> 6;
  const int mBlock = blockIdx.x / nGroups;
  const int nGrp = blockIdx.x % nGroups;
  const int n0 = nGrp * 64;
  const int sStride = K + 8;

  // Cooperative stage of Wt[n0..n0+63][0..K) into LDS (group rows contiguous).
  const int rowChunks = K >> 3;  // v8bf chunks per row
  for (int i = tid; i < 64 * rowChunks; i += 256) {
    const int r = i / rowChunks, c = i - r * rowChunks;
    *(v8bf*)(sWt + r * sStride + c * 8) =
        *(const v8bf*)(Wt + (size_t)(n0 + r) * K + c * 8);
  }
  __syncthreads();

  const int mTile = mBlock * 8 + wave;
  const int half = lane >> 4;   // which half-wave (K-slab select)
  const int lidx = lane & 15;   // row (A) / column (B) index within tile
  const int rowA = mTile * 16 + lidx;

  v8f zero = {};
  v8f acc[4] = {zero, zero, zero, zero};

  const bf16* aRow = A + (size_t)rowA * K;
  for (int k = 0; k < K; k += 32) {
    if (k + 32 < K) __builtin_prefetch(aRow + k + 32, 0, 1);  // global_prefetch_b8
    v16bf a = load_frag(aRow + k + half * 8);                 // global stream
#pragma unroll
    for (int t = 0; t < 4; ++t) {
      const bf16* bRow = sWt + (t * 16 + lidx) * sStride + k + half * 8;
      v16bf b = load_frag(bRow);                              // ds_load_b128 x2
      acc[t] = __builtin_amdgcn_wmma_f32_16x16x32_bf16(
          /*neg_a=*/false, a, /*neg_b=*/false, b,
          /*c_mod=*/(short)0, acc[t], /*reuse_a=*/false, /*reuse_b=*/false);
    }
  }

  // C/D layout: VGPR r -> row (half*8 + r), column = lane&15 within tile.
#pragma unroll
  for (int t = 0; t < 4; ++t) {
    const int col = n0 + t * 16 + lidx;
    const float bb = bias[col];
#pragma unroll
    for (int r = 0; r < 8; ++r) {
      const int row = mTile * 16 + half * 8 + r;
      float v = acc[t][r] + bb;
      if (SWISH) v = swishf(v);
      if (OUT_BF16)
        ((bf16*)Out)[(size_t)row * N + col] = (bf16)v;
      else
        ((float*)Out)[(size_t)row * N + col] = v;
    }
  }
}

// Per-edge BatchNorm over (batch, feature): one block per edge, LDS reduce.
__global__ void bn_kernel(bf16* __restrict__ h, const float* __restrict__ gamma,
                          const float* __restrict__ beta, int E, int D, int Bn) {
  const int e = blockIdx.x;
  const int tid = threadIdx.x;
  __shared__ float s1[256];
  __shared__ float s2[256];
  const int total = Bn * D;
  float sum = 0.f, sq = 0.f;
  for (int i = tid; i < total; i += 256) {
    const int b = i / D, f = i - b * D;
    float v = (float)h[((size_t)b * E + e) * D + f];
    sum += v;
    sq += v * v;
  }
  s1[tid] = sum;
  s2[tid] = sq;
  __syncthreads();
  for (int s = 128; s > 0; s >>= 1) {
    if (tid < s) { s1[tid] += s1[tid + s]; s2[tid] += s2[tid + s]; }
    __syncthreads();
  }
  const float mean = s1[0] / total;
  const float var = s2[0] / total - mean * mean;
  const float inv = rsqrtf(var + 1e-5f);
  const float g = gamma[e], bb = beta[e];
  for (int i = tid; i < total; i += 256) {
    const int b = i / D, f = i - b * D;
    const size_t idx = ((size_t)b * E + e) * D + f;
    float v = (float)h[idx];
    h[idx] = (bf16)(g * (v - mean) * inv + bb);
  }
}

// Build edge features h0[b,e,:] = concat(x2[b,tgt[e]], x2[b,src[e]]) in bf16.
__global__ void gather_kernel(const float* __restrict__ x2, const int* __restrict__ ei,
                              bf16* __restrict__ h0, long long total) {
  long long id = (long long)blockIdx.x * blockDim.x + threadIdx.x;
  if (id >= total) return;
  const int f = (int)(id & 127);
  const long long be = id >> 7;
  const int e = (int)(be & (cE - 1));
  const int b = (int)(be >> 14);
  const int node = (f < 64) ? ei[cE + e] : ei[e];  // tgt first, then src
  const int ff = (f < 64) ? f : f - 64;
  h0[id] = (bf16)x2[((size_t)b * cN + node) * cEMB + ff];
}

// Deterministic segment-sum: edges for target group t are contiguous [t*N,(t+1)*N).
__global__ void agg_kernel(const bf16* __restrict__ m, const int* __restrict__ ei,
                           const int* __restrict__ counts, float* __restrict__ agg) {
  const int b = blockIdx.x >> 7;
  const int t = blockIdx.x & 127;
  const int f = threadIdx.x;  // 64
  float s = 0.f;
  const bf16* base = m + ((size_t)b * cE + (size_t)t * cN) * 64 + f;
  for (int j = 0; j < cN; ++j) s += (float)base[(size_t)j * 64];
  const int tg = ei[cE + t * cN];
  const float c = fmaxf((float)counts[tg], 1.f);
  agg[((size_t)b * cN + tg) * 64 + f] = s / c;
}

__global__ void zero_i32_kernel(int* p, int n) {
  int i = blockIdx.x * blockDim.x + threadIdx.x;
  if (i < n) p[i] = 0;
}
__global__ void count_kernel(const int* __restrict__ ei, int* counts, int E) {
  int e = blockIdx.x * blockDim.x + threadIdx.x;
  if (e < E) atomicAdd(&counts[ei[E + e]], 1);
}

// u_in = concat(x1[128], x2[64], agg[64]) -> bf16
__global__ void concat_u_kernel(const float* __restrict__ x1, const float* __restrict__ x2,
                                const float* __restrict__ agg, bf16* __restrict__ uin, int total) {
  int id = blockIdx.x * blockDim.x + threadIdx.x;
  if (id >= total) return;
  const int f = id & 255;
  const int r = id >> 8;
  float v;
  if (f < 128)      v = x1[(size_t)r * 128 + f];
  else if (f < 192) v = x2[(size_t)r * 64 + (f - 128)];
  else              v = agg[(size_t)r * 64 + (f - 192)];
  uin[id] = (bf16)v;
}

__global__ void add_kernel(float* __restrict__ x, const float* __restrict__ u, int n) {
  int i = blockIdx.x * blockDim.x + threadIdx.x;
  if (i < n) x[i] += u[i];
}

__global__ void f2b_kernel(const float* __restrict__ in, bf16* __restrict__ out, int n) {
  int i = blockIdx.x * blockDim.x + threadIdx.x;
  if (i < n) out[i] = (bf16)in[i];
}

// Convert + transpose fp32 W[K,N] -> bf16 Wt[N,K]
__global__ void wconv_kernel(const float* __restrict__ W, bf16* __restrict__ Wt, int K, int N) {
  int id = blockIdx.x * blockDim.x + threadIdx.x;
  if (id >= K * N) return;
  const int n = id % N, k = id / N;
  Wt[(size_t)n * K + k] = (bf16)W[(size_t)k * N + n];
}

// x2 = node_feature[.,2] @ W[2,64] + b
__global__ void emb_kernel(const float* __restrict__ nf, const float* __restrict__ w,
                           const float* __restrict__ b, float* __restrict__ x2, int rows) {
  int id = blockIdx.x * blockDim.x + threadIdx.x;
  if (id >= rows * 64) return;
  const int f = id & 63, r = id >> 6;
  x2[id] = b[f] + nf[r * 2] * w[f] + nf[r * 2 + 1] * w[64 + f];
}

__global__ void pool_kernel(const float* __restrict__ h, float* __restrict__ pooled) {
  const int b = blockIdx.x, f = threadIdx.x;  // 128
  float s = 0.f;
  for (int n = 0; n < cN; ++n) s += h[((size_t)b * cN + n) * 128 + f];
  pooled[b * 128 + f] = s * (1.f / 128.f);
}

__global__ void post_kernel(const float* __restrict__ pooled, const float* __restrict__ w0,
                            const float* __restrict__ b0, const float* __restrict__ w1,
                            const float* __restrict__ b1, float* __restrict__ coeff) {
  const int b = blockIdx.x, t = threadIdx.x;  // 128
  __shared__ float sh[128];
  float a = b0[t];
  for (int f = 0; f < 128; ++f) a += pooled[b * 128 + f] * w0[f * 128 + t];
  sh[t] = swishf(a);
  __syncthreads();
  if (t < 64) {
    float c = b1[t];
    for (int k = 0; k < 128; ++k) c += sh[k] * w1[k * 64 + t];
    coeff[b * 64 + t] = c;
  }
}

// x3[b,w] = sum_{n<64} x1[b,n,w] * coeff[b,n]
__global__ void final_kernel(const float* __restrict__ x1, const float* __restrict__ coeff,
                             float* __restrict__ out) {
  const int b = blockIdx.x, w = threadIdx.x;  // 128
  float s = 0.f;
  for (int n = 0; n < 64; ++n) s += x1[((size_t)b * cN + n) * 128 + w] * coeff[b * 64 + n];
  out[b * 128 + w] = s;
}

// ---------------------------------------------------------------------------
static void gemm(const bf16* A, const bf16* Wt, const float* bias, void* out,
                 int M, int N, int K, bool swish, bool outbf, hipStream_t s) {
  const int blocks = (M >> 7) * (N >> 6);  // block = 128 rows x 64 cols
  dim3 g(blocks), blk(256);
  if (outbf) {
    if (swish) wmma_gemm_kernel<true, true><<<g, blk, 0, s>>>(A, Wt, bias, out, M, N, K);
    else       wmma_gemm_kernel<true, false><<<g, blk, 0, s>>>(A, Wt, bias, out, M, N, K);
  } else {
    if (swish) wmma_gemm_kernel<false, true><<<g, blk, 0, s>>>(A, Wt, bias, out, M, N, K);
    else       wmma_gemm_kernel<false, false><<<g, blk, 0, s>>>(A, Wt, bias, out, M, N, K);
  }
}

extern "C" void kernel_launch(void* const* d_in, const int* in_sizes, int n_in,
                              void* d_out, int out_size, void* d_ws, size_t ws_size,
                              hipStream_t stream) {
  (void)n_in; (void)out_size; (void)ws_size;
  const float* nodef = (const float*)d_in[0];
  const float* vecs  = (const float*)d_in[1];
  const int*   ei    = (const int*)d_in[2];

  // ----- input-leaf index map: handle dict-insertion order OR jax-sorted -----
  int gw_i, gb_i, vw_i[3], vb_i[3];
  int mw_i[4][4], mb_i[4][4], uw_i[4][4], ub_i[4][4];
  int gam_i[4][3], bet_i[4][3];
  int prw_i[4], prb_i[4], pow_i[2], pob_i[2];
  const bool ins = (in_sizes[3] == 2 * cEMB);  // g_emb.w first => insertion order
  if (ins) {
    gw_i = 3; gb_i = 4;
    for (int i = 0; i < 3; ++i) { vw_i[i] = 5 + 2 * i; vb_i[i] = 6 + 2 * i; }
    for (int l = 0; l < 4; ++l) {
      const int L = 11 + 22 * l;
      for (int i = 0; i < 4; ++i) { mw_i[l][i] = L + 2 * i;      mb_i[l][i] = L + 2 * i + 1; }
      for (int j = 0; j < 3; ++j) { gam_i[l][j] = L + 8 + 2 * j; bet_i[l][j] = L + 9 + 2 * j; }
      for (int i = 0; i < 4; ++i) { uw_i[l][i] = L + 14 + 2 * i; ub_i[l][i] = L + 15 + 2 * i; }
    }
    for (int i = 0; i < 4; ++i) { prw_i[i] = 99 + 2 * i;  prb_i[i] = 100 + 2 * i; }
    for (int i = 0; i < 2; ++i) { pow_i[i] = 107 + 2 * i; pob_i[i] = 108 + 2 * i; }
  } else {  // jax tree_flatten: keys sorted alphabetically at every dict level
    gb_i = 3; gw_i = 4;
    for (int l = 0; l < 4; ++l) {
      const int L = 5 + 22 * l;
      for (int j = 0; j < 3; ++j) { bet_i[l][j] = L + 2 * j;     gam_i[l][j] = L + 2 * j + 1; }
      for (int i = 0; i < 4; ++i) { mb_i[l][i] = L + 6 + 2 * i;  mw_i[l][i] = L + 7 + 2 * i; }
      for (int i = 0; i < 4; ++i) { ub_i[l][i] = L + 14 + 2 * i; uw_i[l][i] = L + 15 + 2 * i; }
    }
    for (int i = 0; i < 2; ++i) { pob_i[i] = 93 + 2 * i;  pow_i[i] = 94 + 2 * i; }
    for (int i = 0; i < 4; ++i) { prb_i[i] = 97 + 2 * i;  prw_i[i] = 98 + 2 * i; }
    for (int i = 0; i < 3; ++i) { vb_i[i] = 105 + 2 * i;  vw_i[i] = 106 + 2 * i; }
  }
  auto F = [&](int i) { return (const float*)d_in[i]; };

  // ----- workspace bump allocator -----
  char* wsb = (char*)d_ws;
  size_t off = 0;
  auto alloc = [&](size_t bytes) -> void* {
    void* p = wsb + off;
    off += (bytes + 255) & ~(size_t)255;
    return p;
  };

  const int vecK[3] = {128, 256, 256}, vecN[3] = {256, 256, 128};
  const int msgK[4] = {128, 128, 128, 128}, msgN[4] = {128, 128, 128, 64};
  const int updK[4] = {256, 256, 256, 256}, updN[4] = {256, 256, 256, 64};
  const int preK[4] = {64, 256, 256, 256},  preN[4] = {256, 256, 256, 128};

  bf16* vecWt[3];
  for (int i = 0; i < 3; ++i) vecWt[i] = (bf16*)alloc((size_t)vecK[i] * vecN[i] * 2);
  bf16* msgWt[4][4];
  bf16* updWt[4][4];
  for (int l = 0; l < 4; ++l) {
    for (int i = 0; i < 4; ++i) msgWt[l][i] = (bf16*)alloc((size_t)msgK[i] * msgN[i] * 2);
    for (int i = 0; i < 4; ++i) updWt[l][i] = (bf16*)alloc((size_t)updK[i] * updN[i] * 2);
  }
  bf16* preWt[4];
  for (int i = 0; i < 4; ++i) preWt[i] = (bf16*)alloc((size_t)preK[i] * preN[i] * 2);

  bf16*  vinb  = (bf16*)alloc((size_t)cBN * 128 * 2);
  bf16*  h1b   = (bf16*)alloc((size_t)cBN * 256 * 2);
  bf16*  h2b   = (bf16*)alloc((size_t)cBN * 256 * 2);
  float* x1f   = (float*)alloc((size_t)cBN * 128 * 4);
  float* x2f   = (float*)alloc((size_t)cBN * 64 * 4);
  bf16*  heA   = (bf16*)alloc((size_t)cBE * 128 * 2);  // 64 MiB (L2-resident)
  bf16*  heB   = (bf16*)alloc((size_t)cBE * 128 * 2);  // 64 MiB
  bf16*  mbuf  = heA;                                  // alias: msg output (64-wide)
  float* aggf  = (float*)alloc((size_t)cBN * 64 * 4);
  bf16*  uin   = (bf16*)alloc((size_t)cBN * 256 * 2);
  bf16*  uh1   = (bf16*)alloc((size_t)cBN * 256 * 2);
  bf16*  uh2   = (bf16*)alloc((size_t)cBN * 256 * 2);
  float* uoutf = (float*)alloc((size_t)cBN * 64 * 4);
  bf16*  x2b   = (bf16*)alloc((size_t)cBN * 64 * 2);
  float* hpre  = (float*)alloc((size_t)cBN * 128 * 4);
  float* pooled = (float*)alloc((size_t)cB * 128 * 4);
  float* coeff  = (float*)alloc((size_t)cB * 64 * 4);
  int*   counts = (int*)alloc((size_t)cN * 4);

  // ----- weight convert + transpose to bf16 [N,K] -----
  auto wconv = [&](int widx, bf16* dst, int K, int N) {
    const int tot = K * N;
    wconv_kernel<<<CDIV(tot, 256), 256, 0, stream>>>(F(widx), dst, K, N);
  };
  for (int i = 0; i < 3; ++i) wconv(vw_i[i], vecWt[i], vecK[i], vecN[i]);
  for (int l = 0; l < 4; ++l) {
    for (int i = 0; i < 4; ++i) wconv(mw_i[l][i], msgWt[l][i], msgK[i], msgN[i]);
    for (int i = 0; i < 4; ++i) wconv(uw_i[l][i], updWt[l][i], updK[i], updN[i]);
  }
  for (int i = 0; i < 4; ++i) wconv(prw_i[i], preWt[i], preK[i], preN[i]);

  // ----- embedding + vector MLP -----
  emb_kernel<<<CDIV(cBN * 64, 256), 256, 0, stream>>>(nodef, F(gw_i), F(gb_i), x2f, cBN);
  f2b_kernel<<<CDIV(cBN * 128, 256), 256, 0, stream>>>(vecs, vinb, cBN * 128);
  gemm(vinb, vecWt[0], F(vb_i[0]), h1b, cBN, 256, 128, true, true, stream);
  gemm(h1b, vecWt[1], F(vb_i[1]), h2b, cBN, 256, 256, true, true, stream);
  gemm(h2b, vecWt[2], F(vb_i[2]), x1f, cBN, 128, 256, false, false, stream);

  // ----- in-degree counts -----
  zero_i32_kernel<<<1, 128, 0, stream>>>(counts, cN);
  count_kernel<<<CDIV(cE, 256), 256, 0, stream>>>(ei, counts, cE);

  // ----- message-passing layers -----
  const long long gtot = cBE * 128;
  for (int l = 0; l < 4; ++l) {
    gather_kernel<<<(int)CDIV(gtot, 256), 256, 0, stream>>>(x2f, ei, heA, gtot);
    gemm(heA, msgWt[l][0], F(mb_i[l][0]), heB, (int)cBE, 128, 128, true, true, stream);
    bn_kernel<<<cE, 256, 0, stream>>>(heB, F(gam_i[l][0]), F(bet_i[l][0]), cE, 128, cB);
    gemm(heB, msgWt[l][1], F(mb_i[l][1]), heA, (int)cBE, 128, 128, true, true, stream);
    bn_kernel<<<cE, 256, 0, stream>>>(heA, F(gam_i[l][1]), F(bet_i[l][1]), cE, 128, cB);
    gemm(heA, msgWt[l][2], F(mb_i[l][2]), heB, (int)cBE, 128, 128, true, true, stream);
    bn_kernel<<<cE, 256, 0, stream>>>(heB, F(gam_i[l][2]), F(bet_i[l][2]), cE, 128, cB);
    gemm(heB, msgWt[l][3], F(mb_i[l][3]), mbuf, (int)cBE, 64, 128, true, true, stream);

    agg_kernel<<<cB * cN, 64, 0, stream>>>(mbuf, ei, counts, aggf);
    concat_u_kernel<<<CDIV(cBN * 256, 256), 256, 0, stream>>>(x1f, x2f, aggf, uin, cBN * 256);
    gemm(uin, updWt[l][0], F(ub_i[l][0]), uh1, cBN, 256, 256, true, true, stream);
    gemm(uh1, updWt[l][1], F(ub_i[l][1]), uh2, cBN, 256, 256, true, true, stream);
    gemm(uh2, updWt[l][2], F(ub_i[l][2]), uh1, cBN, 256, 256, true, true, stream);
    gemm(uh1, updWt[l][3], F(ub_i[l][3]), uoutf, cBN, 64, 256, true, false, stream);
    add_kernel<<<CDIV(cBN * 64, 256), 256, 0, stream>>>(x2f, uoutf, cBN * 64);
  }

  // ----- pre-pool MLP, pooling, post head, basis contraction -----
  f2b_kernel<<<CDIV(cBN * 64, 256), 256, 0, stream>>>(x2f, x2b, cBN * 64);
  gemm(x2b, preWt[0], F(prb_i[0]), uh1, cBN, 256, 64, true, true, stream);
  gemm(uh1, preWt[1], F(prb_i[1]), uh2, cBN, 256, 256, true, true, stream);
  gemm(uh2, preWt[2], F(prb_i[2]), uh1, cBN, 256, 256, true, true, stream);
  gemm(uh1, preWt[3], F(prb_i[3]), hpre, cBN, 128, 256, false, false, stream);

  pool_kernel<<<cB, 128, 0, stream>>>(hpre, pooled);
  post_kernel<<<cB, 128, 0, stream>>>(pooled, F(pow_i[0]), F(pob_i[0]),
                                      F(pow_i[1]), F(pob_i[1]), coeff);
  final_kernel<<<cB, 128, 0, stream>>>(x1f, coeff, (float*)d_out);
}
